// ModuleLSTM_91139206021543
// MI455X (gfx1250) — matrix operand order repaired and checked by reading, not compiled, not run on hardware
//
#include <hip/hip_runtime.h>
#include <hip/hip_bf16.h>

// ---------------------------------------------------------------------------
// 2-layer LSTM (B=1024, H=512, T=64, future=8) on gfx1250.
// bf16 WMMA GEMMs (fp32 accumulate), LSTM cell fused into the GEMM epilogue,
// A-tiles moved global->LDS with double-buffered GLOBAL_LOAD_ASYNC_TO_LDS_B128
// (ASYNCcnt pipeline), 3 kernels per timestep.
// ---------------------------------------------------------------------------

typedef __bf16 bf16_t;
typedef bf16_t v16bf __attribute__((ext_vector_type(16)));
typedef float  v8f   __attribute__((ext_vector_type(8)));

#define BB      1024
#define HH      512
#define T_IN    64
#define STEP_IN 24
#define FUT     8
#define TOT_T   (T_IN + FUT)
#define OUTC    24
#define K1      544      // 512 (h) + 32 (x: 24 valid + 8 pad)
#define K2      1024     // 512 (h2) + 512 (h1)

#define MTILE   128      // batch rows per block
#define JTILE   32       // hidden columns per block (=> 4*32 gate columns)
#define KT      32       // K per WMMA step
#define APITCH  40       // LDS row pitch in bf16 (80 B/row: 16B-aligned, low-conflict)

#if __has_builtin(__builtin_amdgcn_s_wait_asynccnt)
#define WAIT_ASYNC(n) __builtin_amdgcn_s_wait_asynccnt(n)
#else
#define WAIT_ASYNC(n) asm volatile("s_wait_asynccnt %0" ::"n"(n) : "memory")
#endif

__device__ __forceinline__ float sigmoidf(float x) {
  return 1.0f / (1.0f + __expf(-x));
}
__device__ __forceinline__ float tanh_fast(float x) {
  x = fminf(15.0f, fmaxf(-15.0f, x));
  float e = __expf(2.0f * x);
  return (e - 1.0f) / (e + 1.0f);
}

union V16 { v16bf v; uint4 q[2]; };

// A-fragment (16x32 bf16) per CDNA5 ISA: lane%16 = M; lanes 0-15 hold
// K=half*8+{0..7} in e0..7 and K=16+half*8+{0..7} in e8..15 (half = lane/16).
// Two 16-byte reads per lane.
__device__ __forceinline__ v16bf load_a_frag_lds(const bf16_t* base, int m0) {
  const int lane = threadIdx.x & 31;
  const int half = lane >> 4;
  const bf16_t* row = base + (m0 + (lane & 15)) * APITCH + half * 8;
  V16 u;
  u.q[0] = *(const uint4*)row;
  u.q[1] = *(const uint4*)(row + 16);
  return u.v;
}

__device__ __forceinline__ v16bf load_a_frag_gmem(const bf16_t* base, int ldm,
                                                  int m0, int k0) {
  const int lane = threadIdx.x & 31;
  const int half = lane >> 4;
  const bf16_t* row = base + (size_t)(m0 + (lane & 15)) * ldm + k0 + half * 8;
  V16 u;
  u.q[0] = *(const uint4*)row;
  u.q[1] = *(const uint4*)(row + 16);
  return u.v;
}

// B-fragment (32x16 bf16): lane%16 = N; e0..7 hold K=half*8+{0..7}, e8..15 hold
// K=16+half*8+{0..7} (matches A) -> two 16-byte reads.
__device__ __forceinline__ v16bf load_b_frag(const bf16_t* w, int ldk,
                                             int n0, int k0) {
  const int lane = threadIdx.x & 31;
  const bf16_t* p = w + (size_t)(n0 + (lane & 15)) * ldk + k0 + ((lane >> 4) << 3);
  V16 u;
  u.q[0] = *(const uint4*)p;
  u.q[1] = *(const uint4*)(p + 16);
  return u.v;
}

// ---------------------------------------------------------------------------
// Fused LSTM layer: gates = [h | x] * Wcat^T + bsum ; cell update in epilogue.
// Block: 128 batch rows x 32 hidden cols (all 4 gates).  Waves: 4 over M x 2
// over hidden-16, so each wave owns i/f/g/o fragments for the same (m, j).
// A tiles: double-buffered LDS fed by global_load_async_to_lds_b128
// (one async b128 per thread per tile), ASYNCcnt-pipelined.
// ---------------------------------------------------------------------------
template <bool XF32>
__global__ __launch_bounds__(256)
void lstm_layer_kernel(const bf16_t* __restrict__ hA,   // [B][512]  A cols [0,512)
                       const void*  __restrict__ xsrc,  // A cols [512, 512+xcols)
                       int xstride, int xcols,
                       const bf16_t* __restrict__ W,    // [2048][ldk] packed
                       int ldk, int ktot,
                       const float* __restrict__ bsum,  // [2048] b_ih+b_hh
                       const float* __restrict__ c_in,  // [B][512] fp32
                       float*       __restrict__ c_out,
                       bf16_t*      __restrict__ h_out) // [B][512] bf16
{
  __shared__ __align__(16) bf16_t atile[2][MTILE * APITCH];
  const int tid  = threadIdx.x;
  const int mB   = blockIdx.x * MTILE;
  const int j0   = blockIdx.y * JTILE;
  const int wave = tid >> 5;
  const int m0   = (wave & 3) * 32;
  const int jb   = j0 + (wave >> 2) * 16;
  const int lane = tid & 31;

  // staging geometry: thread -> (row, 16-col half) of the 128x32 A tile
  const int row = tid >> 1;            // 0..127
  const int cb  = (tid & 1) * 16;      // 0 / 16
  const int rg  = mB + row;
  const uint32_t ldsb0 =
      (uint32_t)(uintptr_t)(&atile[0][row * APITCH + cb]);
  const uint32_t ldsb1 =
      (uint32_t)(uintptr_t)(&atile[1][row * APITCH + cb]);

  const int niter = ktot / KT;

  // Issue one async b128 (8 bf16 row-segment) for tile `ti` into buffer `buf`;
  // layer-1's x tile (f32 source) is staged synchronously instead.
  auto stage = [&](int ti, int buf) {
    const int kt = ti * KT;
    if (!XF32 || kt < HH) {
      const bf16_t* src;
      if (kt < HH) src = hA + (size_t)rg * HH + kt + cb;
      else src = (const bf16_t*)xsrc + (size_t)rg * xstride + (kt - HH) + cb;
      const uint32_t dst = buf ? ldsb1 : ldsb0;
      asm volatile("global_load_async_to_lds_b128 %0, %1, off"
                   :: "v"(dst), "v"(src) : "memory");
    } else {
      // layer-1 x tile: f32 -> bf16 convert + zero pad (kt == HH here)
#pragma unroll
      for (int e = 0; e < 16; ++e) {
        int k2 = cb + e;
        bf16_t v = (k2 < xcols)
            ? (bf16_t)(((const float*)xsrc)[(size_t)rg * xstride + k2])
            : (bf16_t)0.0f;
        atile[buf][row * APITCH + cb + e] = v;
      }
    }
  };

  v8f zero = {};
  v8f acc[2][4];
#pragma unroll
  for (int mi = 0; mi < 2; ++mi)
#pragma unroll
    for (int g = 0; g < 4; ++g) acc[mi][g] = zero;

  stage(0, 0);
  for (int i = 0; i < niter; ++i) {
    const int cur = i & 1;
    bool nextAsync = false;
    if (i + 1 < niter) {
      stage(i + 1, cur ^ 1);
      nextAsync = (!XF32) || ((i + 1) * KT < HH);
    }
    if (nextAsync) WAIT_ASYNC(1); else WAIT_ASYNC(0);
    __syncthreads();

    const bf16_t* at = atile[cur];
    v16bf a0 = load_a_frag_lds(at, m0);
    v16bf a1 = load_a_frag_lds(at, m0 + 16);
    const int kt = i * KT;
#pragma unroll
    for (int g = 0; g < 4; ++g) {
      v16bf b = load_b_frag(W, ldk, g * HH + jb, kt);
      acc[0][g] = __builtin_amdgcn_wmma_f32_16x16x32_bf16(
          false, a0, false, b, (short)0, acc[0][g], false, false);
      acc[1][g] = __builtin_amdgcn_wmma_f32_16x16x32_bf16(
          false, a1, false, b, (short)0, acc[1][g], false, false);
    }
    __syncthreads();
  }

  // Epilogue: LSTM cell update, fully in C fragments.
  // C layout: lane%16 -> N (hidden j), VGPR r -> M = r + 8*(lane>=16).
  const int j  = jb + (lane & 15);
  const float bi = bsum[0 * HH + j];
  const float bf = bsum[1 * HH + j];
  const float bg = bsum[2 * HH + j];
  const float bo = bsum[3 * HH + j];
#pragma unroll
  for (int mi = 0; mi < 2; ++mi) {
#pragma unroll
    for (int r = 0; r < 8; ++r) {
      int m = mB + m0 + mi * 16 + r + ((lane >> 4) << 3);
      float gi = acc[mi][0][r] + bi;
      float gf = acc[mi][1][r] + bf;
      float gg = acc[mi][2][r] + bg;
      float go = acc[mi][3][r] + bo;
      float cp = c_in[(size_t)m * HH + j];
      float c  = sigmoidf(gf) * cp + sigmoidf(gi) * tanh_fast(gg);
      float h  = sigmoidf(go) * tanh_fast(c);
      c_out[(size_t)m * HH + j] = c;
      h_out[(size_t)m * HH + j] = (bf16_t)h;
    }
  }
}

// ---------------------------------------------------------------------------
// Linear head: out[b, tcol..tcol+23] = h2 @ W_lin^T + b_lin  (N padded to 32).
// ---------------------------------------------------------------------------
__global__ __launch_bounds__(256)
void linear_out_kernel(const bf16_t* __restrict__ h2,   // [B][512] bf16
                       const bf16_t* __restrict__ Wl,   // [32][512] bf16 padded
                       const float*  __restrict__ bl,   // [24]
                       float* __restrict__ out,         // [B][out_ld]
                       int out_ld, int tcol)
{
  const int tid  = threadIdx.x;
  const int mB   = blockIdx.x * MTILE;
  const int wave = tid >> 5;
  const int m0   = mB + wave * 16;
  const int lane = tid & 31;

  v8f zero = {};
  v8f acc[2] = {zero, zero};
  for (int kt = 0; kt < HH; kt += KT) {
    v16bf a = load_a_frag_gmem(h2, HH, m0, kt);
#pragma unroll
    for (int ni = 0; ni < 2; ++ni) {
      v16bf b = load_b_frag(Wl, HH, ni * 16, kt);
      acc[ni] = __builtin_amdgcn_wmma_f32_16x16x32_bf16(
          false, a, false, b, (short)0, acc[ni], false, false);
    }
  }
#pragma unroll
  for (int ni = 0; ni < 2; ++ni) {
    int n = ni * 16 + (lane & 15);
    if (n < OUTC) {
      float bb = bl[n];
#pragma unroll
      for (int r = 0; r < 8; ++r) {
        int m = m0 + r + ((lane >> 4) << 3);
        out[(size_t)m * out_ld + tcol + n] = acc[ni][r] + bb;
      }
    }
  }
}

// ---------------------------------------------------------------------------
// One-shot prep: pack fp32 weights -> bf16 concatenated layouts + bias sums.
// ---------------------------------------------------------------------------
__global__ void pack_weights_kernel(const float* __restrict__ Wih1,  // [2048][24]
                                    const float* __restrict__ Whh1,  // [2048][512]
                                    const float* __restrict__ bih1,
                                    const float* __restrict__ bhh1,
                                    const float* __restrict__ Wih2,  // [2048][512]
                                    const float* __restrict__ Whh2,  // [2048][512]
                                    const float* __restrict__ bih2,
                                    const float* __restrict__ bhh2,
                                    const float* __restrict__ Wlin,  // [24][512]
                                    bf16_t* __restrict__ Wcat1,      // [2048][544]
                                    bf16_t* __restrict__ Wcat2,      // [2048][1024]
                                    bf16_t* __restrict__ Wlinp,      // [32][512]
                                    float*  __restrict__ bsum1,      // [2048]
                                    float*  __restrict__ bsum2)      // [2048]
{
  const int N1 = 2048 * K1;
  const int N2 = 2048 * K2;
  const int N3 = 32 * HH;
  const int N4 = 2048;
  const int total = N1 + N2 + N3 + N4 + N4;
  for (int i = blockIdx.x * blockDim.x + threadIdx.x; i < total;
       i += gridDim.x * blockDim.x) {
    if (i < N1) {
      int n = i / K1, k = i % K1;
      float v = (k < HH) ? Whh1[(size_t)n * HH + k]
              : (k < HH + STEP_IN) ? Wih1[(size_t)n * STEP_IN + (k - HH)]
              : 0.0f;
      Wcat1[i] = (bf16_t)v;
    } else if (i < N1 + N2) {
      int j = i - N1;
      int n = j / K2, k = j % K2;
      float v = (k < HH) ? Whh2[(size_t)n * HH + k]
                         : Wih2[(size_t)n * HH + (k - HH)];
      Wcat2[j] = (bf16_t)v;
    } else if (i < N1 + N2 + N3) {
      int j = i - N1 - N2;
      int n = j / HH, k = j % HH;
      Wlinp[j] = (bf16_t)((n < OUTC) ? Wlin[(size_t)n * HH + k] : 0.0f);
    } else if (i < N1 + N2 + N3 + N4) {
      int j = i - N1 - N2 - N3;
      bsum1[j] = bih1[j] + bhh1[j];
    } else {
      int j = i - N1 - N2 - N3 - N4;
      bsum2[j] = bih2[j] + bhh2[j];
    }
  }
}

__global__ void init_state_kernel(bf16_t* h1, bf16_t* h2, float* c1, float* c2) {
  const int n = BB * HH;
  for (int i = blockIdx.x * blockDim.x + threadIdx.x; i < n;
       i += gridDim.x * blockDim.x) {
    h1[i] = (bf16_t)0.0f;
    h2[i] = (bf16_t)0.0f;
    c1[i] = 0.0f;
    c2[i] = 0.0f;
  }
}

// ---------------------------------------------------------------------------

extern "C" void kernel_launch(void* const* d_in, const int* in_sizes, int n_in,
                              void* d_out, int out_size, void* d_ws, size_t ws_size,
                              hipStream_t stream) {
  (void)in_sizes; (void)n_in; (void)out_size; (void)ws_size;

  const float* input = (const float*)d_in[0];
  const float* W_ih1 = (const float*)d_in[1];
  const float* W_hh1 = (const float*)d_in[2];
  const float* b_ih1 = (const float*)d_in[3];
  const float* b_hh1 = (const float*)d_in[4];
  const float* W_ih2 = (const float*)d_in[5];
  const float* W_hh2 = (const float*)d_in[6];
  const float* b_ih2 = (const float*)d_in[7];
  const float* b_hh2 = (const float*)d_in[8];
  const float* W_lin = (const float*)d_in[9];
  const float* b_lin = (const float*)d_in[10];
  float* out = (float*)d_out;

  // Workspace carve-out.
  char* ws = (char*)d_ws;
  size_t off = 0;
  auto carve = [&](size_t bytes) -> char* {
    char* p = ws + off;
    off += (bytes + 255) & ~(size_t)255;
    return p;
  };
  bf16_t* Wcat1 = (bf16_t*)carve((size_t)2048 * K1 * sizeof(bf16_t));
  bf16_t* Wcat2 = (bf16_t*)carve((size_t)2048 * K2 * sizeof(bf16_t));
  bf16_t* Wlinp = (bf16_t*)carve((size_t)32 * HH * sizeof(bf16_t));
  float*  bsum1 = (float*)carve(2048 * sizeof(float));
  float*  bsum2 = (float*)carve(2048 * sizeof(float));
  bf16_t* h1b[2];
  bf16_t* h2b[2];
  float*  c1b[2];
  float*  c2b[2];
  for (int i = 0; i < 2; ++i) {
    h1b[i] = (bf16_t*)carve((size_t)BB * HH * sizeof(bf16_t));
    h2b[i] = (bf16_t*)carve((size_t)BB * HH * sizeof(bf16_t));
    c1b[i] = (float*)carve((size_t)BB * HH * sizeof(float));
    c2b[i] = (float*)carve((size_t)BB * HH * sizeof(float));
  }

  pack_weights_kernel<<<512, 256, 0, stream>>>(
      W_ih1, W_hh1, b_ih1, b_hh1, W_ih2, W_hh2, b_ih2, b_hh2, W_lin,
      Wcat1, Wcat2, Wlinp, bsum1, bsum2);
  init_state_kernel<<<512, 256, 0, stream>>>(h1b[0], h2b[0], c1b[0], c2b[0]);

  const dim3 grid(BB / MTILE, HH / JTILE);
  for (int t = 0; t < TOT_T; ++t) {
    const int cur = t & 1;
    const int nxt = cur ^ 1;
    const float* x;
    int xld;
    if (t < T_IN) { x = input + (size_t)t * STEP_IN; xld = T_IN * STEP_IN; }
    else          { x = out + (size_t)(t - 1) * OUTC; xld = TOT_T * OUTC; }

    lstm_layer_kernel<true><<<grid, 256, 0, stream>>>(
        h1b[cur], (const void*)x, xld, STEP_IN,
        Wcat1, K1, K1, bsum1, c1b[cur], c1b[nxt], h1b[nxt]);

    lstm_layer_kernel<false><<<grid, 256, 0, stream>>>(
        h2b[cur], (const void*)h1b[nxt], HH, HH,
        Wcat2, K2, K2, bsum2, c2b[cur], c2b[nxt], h2b[nxt]);

    linear_out_kernel<<<BB / MTILE, 256, 0, stream>>>(
        h2b[nxt], Wlinp, b_lin, out, TOT_T * OUTC, t * OUTC);
  }
}
